// APPNP_35021163331771
// MI455X (gfx1250) — matrix operand order repaired and checked by reading, not compiled, not run on hardware
//
#include <hip/hip_runtime.h>
#include <hip/hip_bf16.h>

// ---------------------------------------------------------------------------
// APPNP on MI455X (gfx1250):
//   - 3-layer MLP as bf16 WMMA GEMMs, B panel staged in LDS (320KB/WGP),
//     10 waves/block sharing the panel; B frags batched, A software-pipelined.
//   - K=10 propagation: f32 gather + global_atomic_add_f32, L2-resident.
// ---------------------------------------------------------------------------

typedef __attribute__((ext_vector_type(16))) __bf16 v16bf;
typedef __attribute__((ext_vector_type(8)))  float  v8f;

union BF16x16 {
  v16bf v;
  uint4 u[2];
};

// --------------------------- WMMA GEMM ------------------------------------
// C[M,Nout] = act(A[M,KT] * W[KT,Nout] + bias), weights pre-transposed bf16
// WT[Nout][KT]. One wave per 16-row M tile, 10 waves per block. For each
// 16*NT-column group the block cooperatively stages the B panel in LDS
// (row-padded 16B for bank-conflict-free ds_load_b128 fragment reads).
template <int KT, int NT, bool A_F32, bool RELU, bool OUT_F32>
__global__ __launch_bounds__(320)
void gemm_wmma_bf16(const void* __restrict__ Aptr,
                    const __bf16* __restrict__ WT,
                    const float* __restrict__ bias,
                    void* __restrict__ Out,
                    int M, int Nout) {
  constexpr int WAVES = 10;
  constexpr int ROWS  = NT * 16;          // B rows (columns of C) per group
  constexpr int ROWB  = KT * 2 + 16;      // bytes per LDS row (+16B pad)
  __shared__ __align__(16) unsigned char smemB[ROWS * ROWB];

  const int lane = threadIdx.x & 31;
  const int wv   = threadIdx.x >> 5;
  const int mtiles = M >> 4;
  const int tile   = blockIdx.x * WAVES + wv;
  const bool active = tile < mtiles;
  const int tileC  = active ? tile : (mtiles - 1);   // clamp, no early return

  const int half = lane >> 4;       // which 16-lane half of the wave
  const int l16  = lane & 15;
  const int rowA = tileC * 16 + l16;

  const float*  Af = (const float*)Aptr;
  const __bf16* Ab = (const __bf16*)Aptr;

  // ---- A fragment loader: 16-bit A 16x32 layout:
  //   lane half h, element e: K = (e>>3)*16 + h*8 + (e&7)
  auto loadA = [&](int kb) -> BF16x16 {
    BF16x16 a;
    const int k0 = kb + half * 8;
    if (A_F32) {
      const float* p = Af + (size_t)rowA * KT + k0;
      float4 x0 = *(const float4*)(p);
      float4 x1 = *(const float4*)(p + 4);
      float4 x2 = *(const float4*)(p + 16);
      float4 x3 = *(const float4*)(p + 20);
      float tf[16] = {x0.x, x0.y, x0.z, x0.w, x1.x, x1.y, x1.z, x1.w,
                      x2.x, x2.y, x2.z, x2.w, x3.x, x3.y, x3.z, x3.w};
#pragma unroll
      for (int i = 0; i < 16; ++i) a.v[i] = (__bf16)tf[i];
    } else {
      const __bf16* p = Ab + (size_t)rowA * KT + k0;
      a.u[0] = *(const uint4*)(p);
      a.u[1] = *(const uint4*)(p + 16);
    }
    return a;
  };

  const int ngroups = Nout / ROWS;
  for (int g = 0; g < ngroups; ++g) {
    const int nBase = g * ROWS;
    // ---- cooperative B panel fill: global (coalesced rows) -> padded LDS
    constexpr int CHUNKS = ROWS * (KT / 8);          // 16B chunks
    for (int i = threadIdx.x; i < CHUNKS; i += blockDim.x) {
      const int r  = i / (KT / 8);
      const int c8 = i - r * (KT / 8);
      const uint4 val = *(const uint4*)(WT + (size_t)(nBase + r) * KT + c8 * 8);
      *(uint4*)(smemB + (size_t)r * ROWB + (size_t)c8 * 16) = val;
    }
    __syncthreads();

    v8f acc[NT] = {};
    BF16x16 a = loadA(0);                 // software-pipelined A fragment
#pragma unroll 4
    for (int kb = 0; kb < KT; kb += 32) {
      // ---- all NT B fragments first (distinct regs -> one clause, one wait)
      //   lane holds column N = l16 of the tile, K = (e>>3)*16 + half*8 + (e&7).
      BF16x16 b[NT];
#pragma unroll
      for (int t = 0; t < NT; ++t) {
        const unsigned char* bp =
            smemB + (size_t)(t * 16 + l16) * ROWB + (size_t)(kb + half * 8) * 2;
        b[t].u[0] = *(const uint4*)(bp);
        b[t].u[1] = *(const uint4*)(bp + 32);
      }
      // ---- prefetch next A while the WMMAs below run
      BF16x16 a_next;
      if (kb + 32 < KT) a_next = loadA(kb + 32);
      // ---- back-to-back WMMAs (independent accumulators)
#pragma unroll
      for (int t = 0; t < NT; ++t) {
        acc[t] = __builtin_amdgcn_wmma_f32_16x16x32_bf16(
            false, a.v, false, b[t].v, (short)0, acc[t], false, false);
      }
      a = a_next;
    }
    __syncthreads();   // all waves done reading before next group's fill

    // ---- epilogue: C/D layout: VGPR r, lane: M = r + half*8, N = l16
    if (active) {
#pragma unroll
      for (int t = 0; t < NT; ++t) {
        const int n  = nBase + t * 16 + l16;
        const float bv = bias[n];
#pragma unroll
        for (int r = 0; r < 8; ++r) {
          float v = acc[t][r] + bv;
          if (RELU) v = fmaxf(v, 0.0f);
          const int m = tile * 16 + half * 8 + r;
          if (OUT_F32) ((float*)Out)[(size_t)m * Nout + n] = v;
          else ((__bf16*)Out)[(size_t)m * Nout + n] = (__bf16)v;
        }
      }
    }
  }
}

// ----------------------- small helper kernels ------------------------------
__global__ void k_transpose_bf16(const float* __restrict__ W,
                                 __bf16* __restrict__ WT, int K, int Nout) {
  int i = blockIdx.x * blockDim.x + threadIdx.x;
  if (i >= K * Nout) return;
  int k = i / Nout, n = i - k * Nout;
  WT[(size_t)n * K + k] = (__bf16)W[i];
}

__global__ void k_zero(float* __restrict__ p, int n) {
  int i = blockIdx.x * blockDim.x + threadIdx.x;
  if (i < n) p[i] = 0.0f;
}

__global__ void k_deg(const int* __restrict__ src, const int* __restrict__ dst,
                      float* __restrict__ dout, float* __restrict__ din, int E) {
  int e = blockIdx.x * blockDim.x + threadIdx.x;
  if (e >= E) return;
  unsafeAtomicAdd(&dout[src[e]], 1.0f);
  unsafeAtomicAdd(&din[dst[e]], 1.0f);
}

__global__ void k_norm(float* __restrict__ d, int n) {
  int i = blockIdx.x * blockDim.x + threadIdx.x;
  if (i < n) d[i] = __frsqrt_rn(fmaxf(d[i], 1.0f));
}

// hs = h * norm_out[row], and zero the accumulation buffer in the same pass
__global__ void k_scale_zero(const float* __restrict__ h,
                             const float* __restrict__ norm_out,
                             float* __restrict__ hs, float* __restrict__ acc,
                             int total) {
  int i = blockIdx.x * blockDim.x + threadIdx.x;
  if (i >= total) return;
  acc[i] = 0.0f;
  hs[i]  = h[i] * norm_out[(unsigned)i / 48u];
}

// 16 lanes per edge, 3 channels each (48 = 16*3): coalesced row gather+scatter
__global__ void k_scatter(const int* __restrict__ src, const int* __restrict__ dst,
                          const float* __restrict__ hs, float* __restrict__ acc,
                          int E) {
  int tid = blockIdx.x * blockDim.x + threadIdx.x;
  int e = tid >> 4;
  if (e >= E) return;
  int c = (tid & 15) * 3;
  const float* sp = hs + (size_t)src[e] * 48 + c;
  float* dp = acc + (size_t)dst[e] * 48 + c;
  unsafeAtomicAdd(dp + 0, sp[0]);
  unsafeAtomicAdd(dp + 1, sp[1]);
  unsafeAtomicAdd(dp + 2, sp[2]);
}

// out = (1-alpha) * acc * norm_in[row] + alpha * h0
__global__ void k_finalize(const float* __restrict__ acc,
                           const float* __restrict__ norm_in,
                           const float* __restrict__ h0,
                           float* __restrict__ out, int total) {
  int i = blockIdx.x * blockDim.x + threadIdx.x;
  if (i >= total) return;
  out[i] = 0.9f * acc[i] * norm_in[(unsigned)i / 48u] + 0.1f * h0[i];
}

// --------------------------- orchestration ---------------------------------
extern "C" void kernel_launch(void* const* d_in, const int* in_sizes, int n_in,
                              void* d_out, int out_size, void* d_ws, size_t ws_size,
                              hipStream_t stream) {
  const float* features = (const float*)d_in[0];
  const int*   eidx     = (const int*)d_in[1];
  const float* W0 = (const float*)d_in[2];
  const float* b0 = (const float*)d_in[3];
  const float* W1 = (const float*)d_in[4];
  const float* b1 = (const float*)d_in[5];
  const float* W2 = (const float*)d_in[6];
  const float* b2 = (const float*)d_in[7];

  const int N = in_sizes[0] / 512;
  const int E = in_sizes[1] / 2;
  const int* src = eidx;
  const int* dst = eidx + E;

  char* ws = (char*)d_ws;
  size_t off = 0;
  auto take = [&](size_t bytes) -> char* {
    char* p = ws + off;
    off = (off + bytes + 255) & ~(size_t)255;
    return p;
  };
  __bf16* WT0  = (__bf16*)take((size_t)512 * 256 * 2);
  __bf16* WT1  = (__bf16*)take((size_t)256 * 256 * 2);
  __bf16* WT2  = (__bf16*)take((size_t)256 * 48 * 2);
  __bf16* h1b  = (__bf16*)take((size_t)N * 256 * 2);
  __bf16* h2b  = (__bf16*)take((size_t)N * 256 * 2);
  float*  h0f  = (float*)take((size_t)N * 48 * 4);
  float*  bufA = (float*)take((size_t)N * 48 * 4);
  float*  bufB = (float*)take((size_t)N * 48 * 4);
  float*  accb = (float*)take((size_t)N * 48 * 4);
  float*  deg  = (float*)take((size_t)2 * N * 4);   // [norm_out | norm_in]
  float* norm_out = deg;
  float* norm_in  = deg + N;

  // ---- weight transpose + f32->bf16 convert
  k_transpose_bf16<<<(512 * 256 + 255) / 256, 256, 0, stream>>>(W0, WT0, 512, 256);
  k_transpose_bf16<<<(256 * 256 + 255) / 256, 256, 0, stream>>>(W1, WT1, 256, 256);
  k_transpose_bf16<<<(256 * 48  + 255) / 256, 256, 0, stream>>>(W2, WT2, 256, 48);

  // ---- 3-layer MLP (WMMA bf16, f32 accumulate), 10 waves/block
  const int mtiles  = N / 16;               // 6250 (N = 100000)
  const int gblocks = (mtiles + 9) / 10;    // 625, exact
  gemm_wmma_bf16<512, 4, true,  true,  false><<<gblocks, 320, 0, stream>>>(
      features, WT0, b0, h1b, N, 256);
  gemm_wmma_bf16<256, 4, false, true,  false><<<gblocks, 320, 0, stream>>>(
      h1b, WT1, b1, h2b, N, 256);
  gemm_wmma_bf16<256, 3, false, false, true ><<<gblocks, 320, 0, stream>>>(
      h2b, WT2, b2, h0f, N, 48);

  // ---- degree norms
  k_zero<<<(2 * N + 255) / 256, 256, 0, stream>>>(deg, 2 * N);
  k_deg<<<(E + 255) / 256, 256, 0, stream>>>(src, dst, norm_out, norm_in, E);
  k_norm<<<(2 * N + 255) / 256, 256, 0, stream>>>(deg, 2 * N);

  // ---- K=10 APPNP propagation (h fits in L2: atomics stay on-chip)
  const int total = N * 48;
  const int eb = (int)(((size_t)E * 16 + 255) / 256);
  const float* hcur = h0f;
  for (int k = 0; k < 10; ++k) {
    float* hsbuf = (k & 1) ? bufB : bufA;
    float* hnext = (k == 9) ? (float*)d_out : hsbuf;
    k_scale_zero<<<(total + 255) / 256, 256, 0, stream>>>(hcur, norm_out, hsbuf, accb, total);
    k_scatter<<<eb, 256, 0, stream>>>(src, dst, hsbuf, accb, E);
    k_finalize<<<(total + 255) / 256, 256, 0, stream>>>(accb, norm_in, h0f, hnext, total);
    hcur = hnext;
  }
}
